// ArcFace_21036749816507
// MI455X (gfx1250) — compile-verified, matrix operand order checked by reading, not run on hardware
//
#include <hip/hip_runtime.h>
#include <math.h>

typedef __attribute__((ext_vector_type(16))) _Float16 v16h;
typedef __attribute__((ext_vector_type(8)))  _Float16 v8h;
typedef __attribute__((ext_vector_type(4)))  _Float16 v4h;
typedef __attribute__((ext_vector_type(8)))  float    v8f;
typedef __attribute__((ext_vector_type(4)))  float    v4f;

#define DIMS     512
#define NCLASS   100000
#define NBATCH   1024
#define SCALE    20.0f
#define COS_M    0.8775825618903728f   // cos(0.5)
#define SIN_M    0.479425538604203f    // sin(0.5)
#define LDSROW   520   // 512 + 8 halves pad -> 1040B row stride, spreads DS banks

// ---------------------------------------------------------------------------
// Pass 1/2: L2-normalize rows of a [nrows x 512] fp32 matrix, emit f16 rows.
// One wave (32 lanes) per row; each lane owns 16 columns (4x float4).
// ---------------------------------------------------------------------------
__global__ __launch_bounds__(256) void arcface_norm_f16(
    const float* __restrict__ src, _Float16* __restrict__ dst, int nrows) {
  const int wave = threadIdx.x >> 5;
  const int lane = threadIdx.x & 31;
  const int row  = blockIdx.x * 8 + wave;
  if (row >= nrows) return;

  const float* r = src + (size_t)row * DIMS;
  float vals[16];
  float ss = 0.0f;
#pragma unroll
  for (int c = 0; c < 4; ++c) {
    v4f x = *(const v4f*)(r + c * 128 + lane * 4);
    vals[c * 4 + 0] = x.x; vals[c * 4 + 1] = x.y;
    vals[c * 4 + 2] = x.z; vals[c * 4 + 3] = x.w;
    ss += x.x * x.x + x.y * x.y + x.z * x.z + x.w * x.w;
  }
  // wave32 butterfly reduction
#pragma unroll
  for (int off = 16; off > 0; off >>= 1)
    ss += __shfl_xor(ss, off, 32);

  const float inv = 1.0f / fmaxf(sqrtf(ss), 1e-12f);

  _Float16* o = dst + (size_t)row * DIMS;
#pragma unroll
  for (int c = 0; c < 4; ++c) {
    v4h h;
    h.x = (_Float16)(vals[c * 4 + 0] * inv);
    h.y = (_Float16)(vals[c * 4 + 1] * inv);
    h.z = (_Float16)(vals[c * 4 + 2] * inv);
    h.w = (_Float16)(vals[c * 4 + 3] * inv);
    *(v4h*)(o + c * 128 + lane * 4) = h;
  }
}

// ---------------------------------------------------------------------------
// Pass 3: cos_theta = f16(f_n) x f16(w_n)^T via v_wmma_f32_16x16x32_f16.
// Block = 256 threads = 8 waves; block stages a 32-row w-tile in LDS.
// Each wave computes a 32(M)x32(N) output tile: 2 A-tiles x 2 B-tiles ->
// 4 f32 accumulators, so every fragment load feeds 2 WMMAs. K fully unrolled
// (16 steps of K=32 -> 64 v_wmma per wave).
//
// Fused ArcFace epilogue without transcendentals:
//   non-target: cos(acos(x)) = x            -> out = SCALE * clip(x)
//   target:     cos(acos(x)+m) = x*cos(m) - sin(m)*sqrt(1-x^2)
//
// grid = (100000/32, 1024/256) = (3125, 4)
// ---------------------------------------------------------------------------
__global__ __launch_bounds__(256) void arcface_gemm_wmma(
    const _Float16* __restrict__ fh,   // [1024 x 512] normalized f16
    const _Float16* __restrict__ wh,   // [100000 x 512] normalized f16
    const int*      __restrict__ targets,
    float*          __restrict__ out)  // [1024 x 100000]
{
  __shared__ _Float16 smem[32 * LDSROW];

  const int colBase = blockIdx.x * 32;

  // Cooperative stage: 32 w-rows x 512 halves = 2048 x 16B chunks.
  for (int c = threadIdx.x; c < 2048; c += 256) {
    const int wrow = c >> 6;          // 0..31
    const int ko   = (c & 63) * 8;    // 0..504 step 8
    *(v8h*)&smem[wrow * LDSROW + ko] =
        *(const v8h*)(wh + (size_t)(colBase + wrow) * DIMS + ko);
  }
  __syncthreads();

  const int wave    = threadIdx.x >> 5;
  const int lane    = threadIdx.x & 31;
  const int m       = lane & 15;       // M (A) / N (B,C,D) index per ISA layout
  const int kh      = lane >> 4;       // K-half selector
  const int rowBase = blockIdx.y * 256 + wave * 32;

  const _Float16* arow0 = fh + (size_t)(rowBase + m) * DIMS;
  const _Float16* arow1 = fh + (size_t)(rowBase + 16 + m) * DIMS;
  const _Float16* b0p   = &smem[(m)      * LDSROW];
  const _Float16* b1p   = &smem[(16 + m) * LDSROW];

  v8f acc00 = {};  // (M-tile 0, N-tile 0)
  v8f acc01 = {};  // (M-tile 0, N-tile 1)
  v8f acc10 = {};  // (M-tile 1, N-tile 0)
  v8f acc11 = {};  // (M-tile 1, N-tile 1)

#pragma unroll
  for (int k0 = 0; k0 < DIMS; k0 += 32) {
    const int off = k0 + kh * 8;

    // A fragments: 16x32 f16 (lane m = row; VGPR0-3 <- K[off..off+7],
    // VGPR4-7 <- K[off+16..off+23], per ISA 16-bit A layout).
    v8h a0lo = *(const v8h*)(arow0 + off);
    v8h a0hi = *(const v8h*)(arow0 + off + 16);
    v16h a0  = __builtin_shufflevector(a0lo, a0hi,
                 0,1,2,3,4,5,6,7,8,9,10,11,12,13,14,15);
    v8h a1lo = *(const v8h*)(arow1 + off);
    v8h a1hi = *(const v8h*)(arow1 + off + 16);
    v16h a1  = __builtin_shufflevector(a1lo, a1hi,
                 0,1,2,3,4,5,6,7,8,9,10,11,12,13,14,15);

    // B fragments: 32x16 f16, column n = lane m, same K gather, from LDS.
    v8h b0lo = *(const v8h*)(b0p + off);
    v8h b0hi = *(const v8h*)(b0p + off + 16);
    v16h b0  = __builtin_shufflevector(b0lo, b0hi,
                 0,1,2,3,4,5,6,7,8,9,10,11,12,13,14,15);
    v8h b1lo = *(const v8h*)(b1p + off);
    v8h b1hi = *(const v8h*)(b1p + off + 16);
    v16h b1  = __builtin_shufflevector(b1lo, b1hi,
                 0,1,2,3,4,5,6,7,8,9,10,11,12,13,14,15);

    acc00 = __builtin_amdgcn_wmma_f32_16x16x32_f16(
        false, a0, false, b0, (short)0, acc00, false, false);
    acc01 = __builtin_amdgcn_wmma_f32_16x16x32_f16(
        false, a0, false, b1, (short)0, acc01, false, false);
    acc10 = __builtin_amdgcn_wmma_f32_16x16x32_f16(
        false, a1, false, b0, (short)0, acc10, false, false);
    acc11 = __builtin_amdgcn_wmma_f32_16x16x32_f16(
        false, a1, false, b1, (short)0, acc11, false, false);
  }

  // Epilogue. D layout: VGPR i -> M = i + 8*kh (per half-wave), N = lane&15.
  const int col0 = colBase + m;
  const int col1 = col0 + 16;
#pragma unroll
  for (int i = 0; i < 8; ++i) {
    const int r0   = rowBase + i + 8 * kh;
    const int r1   = r0 + 16;
    const int tgt0 = targets[r0];
    const int tgt1 = targets[r1];

    float x, y;

    x = fminf(fmaxf(acc00[i], -1.0f), 1.0f);
    y = (tgt0 == col0) ? (x * COS_M - SIN_M * sqrtf(fmaxf(1.0f - x * x, 0.0f))) : x;
    out[(size_t)r0 * NCLASS + col0] = y * SCALE;

    x = fminf(fmaxf(acc01[i], -1.0f), 1.0f);
    y = (tgt0 == col1) ? (x * COS_M - SIN_M * sqrtf(fmaxf(1.0f - x * x, 0.0f))) : x;
    out[(size_t)r0 * NCLASS + col1] = y * SCALE;

    x = fminf(fmaxf(acc10[i], -1.0f), 1.0f);
    y = (tgt1 == col0) ? (x * COS_M - SIN_M * sqrtf(fmaxf(1.0f - x * x, 0.0f))) : x;
    out[(size_t)r1 * NCLASS + col0] = y * SCALE;

    x = fminf(fmaxf(acc11[i], -1.0f), 1.0f);
    y = (tgt1 == col1) ? (x * COS_M - SIN_M * sqrtf(fmaxf(1.0f - x * x, 0.0f))) : x;
    out[(size_t)r1 * NCLASS + col1] = y * SCALE;
  }
}

// ---------------------------------------------------------------------------
extern "C" void kernel_launch(void* const* d_in, const int* in_sizes, int n_in,
                              void* d_out, int out_size, void* d_ws, size_t ws_size,
                              hipStream_t stream) {
  const float* features = (const float*)d_in[0];   // [1024 x 512]
  const int*   targets  = (const int*)d_in[1];     // [1024]
  const float* weights  = (const float*)d_in[2];   // [100000 x 512]
  float*       out      = (float*)d_out;           // [1024 x 100000]

  // Workspace layout: f16 normalized features (1 MB), then f16 normalized weights.
  _Float16* fh = (_Float16*)d_ws;
  _Float16* wh = (_Float16*)((char*)d_ws + (size_t)NBATCH * DIMS * sizeof(_Float16));

  // Normalize features: 1024 rows, 8 rows/block.
  arcface_norm_f16<<<NBATCH / 8, 256, 0, stream>>>(features, fh, NBATCH);
  // Normalize weights: 100000 rows.
  arcface_norm_f16<<<(NCLASS + 7) / 8, 256, 0, stream>>>(weights, wh, NCLASS);

  // GEMM + fused ArcFace epilogue.
  dim3 grid(NCLASS / 32, NBATCH / 256);
  arcface_gemm_wmma<<<grid, 256, 0, stream>>>(fh, wh, targets, out);
}
